// RegionAttention_49349174231397
// MI455X (gfx1250) — matrix operand order repaired and checked by reading, not compiled, not run on hardware
//
#include <hip/hip_runtime.h>
#include <hip/hip_bf16.h>

// ---------------------------------------------------------------------------
// RegionAttention for MI455X (gfx1250, wave32, WMMA).
//
// Problem sizes: B=64, H=30, W=60, D=64, M = B*H*W = 115200.
// GEMM qk = x @ qk_w.T (+bias): M=115200, N=128, K=64 -> ~1.9 GFLOP.
// Memory: ~120 MB total traffic -> ~5 us @ 23.3 TB/s; compute is free on the
// matrix units with V_WMMA_F32_16X16X32_F16 (f16 in, f32 accumulate).
//
// Pooling trick: mean_region(x@Wk^T + b) == (sum_region x) @ Wk^T / count + b,
// so k is never materialized. 3 kernels:
//   K1 pool_xsum   : xsum[bin][k] = fixed-order sum of x rows in bin (4864 bins)
//   K2 pool_gemm   : pooled = xsum @ Wk^T * (1/count) + bias   (WMMA)
//   K3 score_mask  : recompute q tile with WMMA -> LDS -> per-pixel dots,
//                    3->1 linear, sigmoid, Gumbel-softmax, write mask.
// Deterministic (no atomics), ws usage = 2 * 4864 * 64 floats = 2.5 MB,
// fully overwritten each launch.
// ---------------------------------------------------------------------------

typedef __attribute__((ext_vector_type(16))) _Float16 v16h;
typedef __attribute__((ext_vector_type(8)))  float    v8f;

#define BATCH 64
#define HH 30
#define WW 60
#define DD 64
#define HW 1800
#define MM 115200
// bins: pool0 (5x10)=50/batch, pool1 (3x6)=18/batch, pool2 (2x4)=8/batch
#define BIN0 0
#define BIN1 3200      // 64*50
#define BIN2 4352      // +64*18
#define NBIN 4864      // +64*8
#define XSUM_F (NBIN * DD)      // floats in xsum region
#define POOLED_OFF XSUM_F       // pooled region starts after xsum

// --------------------------- K1: per-bin x sums -----------------------------
__global__ __launch_bounds__(256) void pool_xsum_kernel(
    const float* __restrict__ x, float* __restrict__ xsum) {
  int tid = blockIdx.x * 256 + threadIdx.x;
  if (tid >= NBIN * DD) return;
  int bin = tid >> 6;
  int k   = tid & 63;
  int b, h0, w0, kh, kw;
  if (bin < BIN1) {                 // pool (5,10): region 6x6
    int r = bin;        b = r / 50; r -= b * 50;
    int r0 = r / 10, r1 = r - r0 * 10;
    kh = 6;  kw = 6;  h0 = r0 * 6;  w0 = r1 * 6;
  } else if (bin < BIN2) {          // pool (3,6): region 10x10
    int r = bin - BIN1; b = r / 18; r -= b * 18;
    int r0 = r / 6, r1 = r - r0 * 6;
    kh = 10; kw = 10; h0 = r0 * 10; w0 = r1 * 10;
  } else {                          // pool (2,4): region 15x15
    int r = bin - BIN2; b = r / 8;  r -= b * 8;
    int r0 = r / 4, r1 = r - r0 * 4;
    kh = 15; kw = 15; h0 = r0 * 15; w0 = r1 * 15;
  }
  const float* base = x + (((b * HH + h0) * WW) + w0) * DD + k;
  float s = 0.f;
  for (int i = 0; i < kh; ++i)
    for (int j = 0; j < kw; ++j)
      s += base[(i * WW + j) * DD];   // lanes -> consecutive k: coalesced
  xsum[tid] = s;
}

// ---------------- WMMA operand packing helpers (documented layouts) ---------
// A (16x32 f16, MxK): lane l holds row M=l%16; element i=(2d+j):
//   K = (d<4?0:16) + (l/16)*8 + 2*(d&3) + j
// B (32x16 f16, KxN): lane l holds col N=l%16; element i: K = (i<8?0:16) + (l/16)*8 + (i&7)
// C/D (16x16 f32):    lane l holds col N=l%16; VGPR v -> row M = v + 8*(l/16)
__device__ inline void pack_a_f32(const float* __restrict__ row,
                                  int lh, v16h& a0, v16h& a1) {
#pragma unroll
  for (int i = 0; i < 16; ++i) {
    int d = i >> 1, j = i & 1;
    int kl = ((d < 4) ? 0 : 16) + lh * 8 + ((d & 3) << 1) + j;
    a0[i] = (_Float16)row[kl];
    a1[i] = (_Float16)row[32 + kl];
  }
}
__device__ inline void pack_b_w(const float* __restrict__ qk_w, int n, int lh,
                                v16h& b0, v16h& b1) {
  const float* wr = qk_w + n * DD;
#pragma unroll
  for (int i = 0; i < 16; ++i) {
    int k0 = ((i < 8) ? 0 : 16) + lh * 8 + (i & 7);
    b0[i] = (_Float16)wr[k0];
    b1[i] = (_Float16)wr[32 + k0];
  }
}

// --------------------- K2: pooled = xsum @ Wk^T /cnt + bias -----------------
__global__ __launch_bounds__(256) void pool_gemm_kernel(
    const float* __restrict__ xsum, const float* __restrict__ qk_w,
    const float* __restrict__ qk_b, float* __restrict__ pooled) {
  int lane = threadIdx.x & 31, wave = threadIdx.x >> 5;
  int lh = lane >> 4, lm = lane & 15;
  int m0 = (blockIdx.x * 8 + wave) * 16;          // 38 blocks * 8 waves * 16 = 4864
  v16h a0, a1;
  pack_a_f32(xsum + (m0 + lm) * DD, lh, a0, a1);
  for (int t = 0; t < 4; ++t) {                   // d columns 16t..16t+15
    int dcol = 16 * t + lm;
    int n = DD + dcol;                            // k-projection rows of qk_w
    v16h b0, b1;
    pack_b_w(qk_w, n, lh, b0, b1);
    v8f c = {};
    c = __builtin_amdgcn_wmma_f32_16x16x32_f16(false, a0, false, b0,
                                               (short)0, c, false, false);
    c = __builtin_amdgcn_wmma_f32_16x16x32_f16(false, a1, false, b1,
                                               (short)0, c, false, false);
    float bias = qk_b[n];
#pragma unroll
    for (int v = 0; v < 8; ++v) {
      int m = m0 + v + 8 * lh;
      float inv = (m < BIN1) ? (1.f / 36.f)
                : (m < BIN2) ? (1.f / 100.f)
                             : (1.f / 225.f);
      pooled[m * DD + dcol] = c[v] * inv + bias;  // region mean of k
    }
  }
}

// ------ K3: q tile via WMMA -> LDS -> scores -> Gumbel-softmax mask ---------
__global__ __launch_bounds__(256) void score_mask_kernel(
    const float* __restrict__ x, const float* __restrict__ qk_w,
    const float* __restrict__ qk_b, const float* __restrict__ score_w,
    const float* __restrict__ score_b, const float* __restrict__ noise_x,
    const float* __restrict__ noise_r, const float* __restrict__ pooled,
    float* __restrict__ out) {
  __shared__ float qls[8][16][68];                // 68-stride pad: no bank conflicts
  int lane = threadIdx.x & 31, wave = threadIdx.x >> 5;
  int lh = lane >> 4, lm = lane & 15;
  int m0 = (blockIdx.x * 8 + wave) * 16;          // 900 blocks * 8 * 16 = 115200

  v16h a0, a1;
  pack_a_f32(x + (m0 + lm) * DD, lh, a0, a1);
  for (int t = 0; t < 4; ++t) {                   // q columns 16t..16t+15
    int n = 16 * t + lm;
    v16h b0, b1;
    pack_b_w(qk_w, n, lh, b0, b1);
    v8f c = {};
    c = __builtin_amdgcn_wmma_f32_16x16x32_f16(false, a0, false, b0,
                                               (short)0, c, false, false);
    c = __builtin_amdgcn_wmma_f32_16x16x32_f16(false, a1, false, b1,
                                               (short)0, c, false, false);
    float bias = qk_b[n];
#pragma unroll
    for (int v = 0; v < 8; ++v)
      qls[wave][v + 8 * lh][n] = c[v] + bias;
  }
  __syncthreads();

  if (lane < 16) {
    int m = m0 + lane;
    int b = m / HW;
    int rem = m - b * HW;
    int h = rem / WW;
    int w = rem - h * WW;
    const float* q  = qls[wave][lane];
    const float* p0 = pooled + (BIN0 + b * 50 + (h / 6)  * 10 + (w / 6))  * DD;
    const float* p1 = pooled + (BIN1 + b * 18 + (h / 10) * 6  + (w / 10)) * DD;
    const float* p2 = pooled + (BIN2 + b * 8  + (h / 15) * 4  + (w / 15)) * DD;
    float d0 = 0.f, d1 = 0.f, d2 = 0.f;
#pragma unroll 8
    for (int n = 0; n < DD; ++n) {
      float qa = q[n];
      d0 += qa * p0[n];
      d1 += qa * p1[n];
      d2 += qa * p2[n];
    }
    // scores * D^-0.5, 3->1 linear
    float attn = 0.125f * (d0 * score_w[0] + d1 * score_w[1] + d2 * score_w[2])
               + score_b[0];
    float as = 1.f / (1.f + __builtin_expf(-attn));
    float gx = __builtin_logf(as + 1e-8f);
    float gr = __builtin_logf(1.f - as + 1e-8f);
    float nx = -__builtin_logf(-__builtin_logf(noise_x[m] + 1e-8f) + 1e-8f);
    float nr = -__builtin_logf(-__builtin_logf(noise_r[m] + 1e-8f) + 1e-8f);
    const float invT = 1.f / (0.03f + 1e-8f);
    gx = (gx + nx) * invT;
    gr = (gr + nr) * invT;
    // softmax([gx, gr])[0]
    out[m] = 1.f / (1.f + __builtin_expf(gr - gx));
  }
}

// ------------------------------- launcher -----------------------------------
extern "C" void kernel_launch(void* const* d_in, const int* in_sizes, int n_in,
                              void* d_out, int out_size, void* d_ws,
                              size_t ws_size, hipStream_t stream) {
  const float* x       = (const float*)d_in[0];
  const float* qk_w    = (const float*)d_in[1];
  const float* qk_b    = (const float*)d_in[2];
  const float* score_w = (const float*)d_in[3];
  const float* score_b = (const float*)d_in[4];
  const float* noise_x = (const float*)d_in[5];
  const float* noise_r = (const float*)d_in[6];
  float* out    = (float*)d_out;
  float* xsum   = (float*)d_ws;            // 4864*64 floats
  float* pooled = xsum + POOLED_OFF;       // 4864*64 floats (2.5 MB total)

  pool_xsum_kernel<<<(NBIN * DD + 255) / 256, 256, 0, stream>>>(x, xsum);
  pool_gemm_kernel<<<NBIN / (8 * 16), 256, 0, stream>>>(xsum, qk_w, qk_b,
                                                        pooled);
  score_mask_kernel<<<MM / (8 * 16), 256, 0, stream>>>(
      x, qk_w, qk_b, score_w, score_b, noise_x, noise_r, pooled, out);
}
